// Arxiv_GCN_5471788335235
// MI455X (gfx1250) — matrix operand order repaired and checked by reading, not compiled, not run on hardware
//
#include <hip/hip_runtime.h>

#define N_NODES 169343
#define NPAD    169344   // multiple of 16
#define BN_EPS  1e-5f

typedef long long i64;
typedef __attribute__((ext_vector_type(16))) __bf16 v16bf;
typedef __attribute__((ext_vector_type(8)))  float  v8f;

union BfFrag { v16bf v; uint4 q[2]; };
union BfPack4 { __bf16 b[4]; uint2 u; };

__device__ __forceinline__ __bf16 f2bf(float f) {
    unsigned u = __builtin_bit_cast(unsigned, f);
    u += 0x7FFFu + ((u >> 16) & 1u);              // round-to-nearest-even
    unsigned short h = (unsigned short)(u >> 16);
    return __builtin_bit_cast(__bf16, h);
}

// ---------------------------------------------------------------- utilities
__global__ void fill_kernel(float* __restrict__ p, float v, long n) {
    long i = (long)blockIdx.x * blockDim.x + threadIdx.x;
    long stride = (long)gridDim.x * blockDim.x;
    for (; i < n; i += stride) p[i] = v;
}

__global__ void degree_kernel(const i64* __restrict__ ei, float* __restrict__ deg, long E) {
    long e = (long)blockIdx.x * blockDim.x + threadIdx.x;
    if (e < E) atomicAdd(&deg[ei[E + e]], 1.0f);
}

__global__ void edgenorm_kernel(const i64* __restrict__ ei, const float* __restrict__ deg,
                                float* __restrict__ en, long E) {
    long e = (long)blockIdx.x * blockDim.x + threadIdx.x;
    if (e < E) en[e] = rsqrtf(deg[ei[e]]) * rsqrtf(deg[ei[E + e]]);
}

// streaming f32 -> bf16 (n must be a multiple of 4)
__global__ void cvt_bf16_kernel(const float* __restrict__ src, __bf16* __restrict__ dst, long n) {
    long i = ((long)blockIdx.x * blockDim.x + threadIdx.x) * 4;
    if (i >= n) return;
    float4 v = *(const float4*)(src + i);
    BfPack4 p;
    p.b[0] = f2bf(v.x); p.b[1] = f2bf(v.y); p.b[2] = f2bf(v.z); p.b[3] = f2bf(v.w);
    *(uint2*)(dst + i) = p.u;
}

// W[K,NV] (row-major f32) -> Wt[NN,K] (col-major bf16), cols >= NV zero-padded
__global__ void wtrans_kernel(const float* __restrict__ W, __bf16* __restrict__ Wt,
                              int K, int NV, int NN) {
    int idx = blockIdx.x * blockDim.x + threadIdx.x;
    if (idx >= NN * K) return;
    int col = idx / K, k = idx % K;
    float v = (col < NV) ? W[(long)k * NV + col] : 0.0f;
    Wt[idx] = f2bf(v);
}

// ---------------------------------------------------------------- GEMM (bf16 WMMA, f32 accum)
// C[NPAD, NN] = A[M, K] @ Wt[NN, K]^T ; each wave: one 16-row M-tile x NT 16-col N-tiles
template<int K, int NN, int NT>
__global__ void gemm_wmma_kernel(const __bf16* __restrict__ A, const __bf16* __restrict__ Wt,
                                 float* __restrict__ C, int Mvalid, int totalGroups) {
    const int NG = NN / (16 * NT);               // N-tile groups per row-tile
    int wave = threadIdx.x >> 5;
    int lane = threadIdx.x & 31;
    int grp = blockIdx.x * (blockDim.x >> 5) + wave;
    if (grp >= totalGroups) return;
    int mt = grp / NG, ng = grp % NG;
    int half = lane >> 4;                        // 0: lanes 0-15, 1: lanes 16-31
    int l16  = lane & 15;
    int arow = mt * 16 + l16;
    bool arowOK = (arow < Mvalid);

    v8f acc[NT];
    #pragma unroll
    for (int t = 0; t < NT; t++) acc[t] = (v8f){};

    #pragma unroll
    for (int k0 = 0; k0 < K; k0 += 32) {
        // A fragment: row = arow, K chunks [c..c+7] and [c+16..c+23], c = 8*half
        BfFrag a;
        if (arowOK) {
            const uint4* ap = (const uint4*)(A + (long)arow * K + k0 + half * 8);
            a.q[0] = ap[0];                      // K + c .. c+7
            a.q[1] = ap[2];                      // K + c+16 .. c+23
        } else {
            a.q[0] = make_uint4(0u, 0u, 0u, 0u);
            a.q[1] = make_uint4(0u, 0u, 0u, 0u);
        }
        // B fragments: column = col, K rows [c..c+7] and [c+16..c+23], c = 8*half (same as A)
        #pragma unroll
        for (int t = 0; t < NT; t++) {
            int col = (ng * NT + t) * 16 + l16;
            const uint4* bp = (const uint4*)(Wt + (long)col * K + k0 + half * 8);
            BfFrag b;
            b.q[0] = bp[0];
            b.q[1] = bp[2];
            acc[t] = __builtin_amdgcn_wmma_f32_16x16x32_bf16(false, a.v, false, b.v,
                                                             (short)0, acc[t], false, false);
        }
    }
    // C: VGPR v -> row mt*16 + 8*half + v, col
    int crow = mt * 16 + half * 8;
    #pragma unroll
    for (int t = 0; t < NT; t++) {
        int ccol = (ng * NT + t) * 16 + l16;
        #pragma unroll
        for (int v = 0; v < 8; v++) C[(long)(crow + v) * NN + ccol] = acc[t][v];
    }
}

// ---------------------------------------------------------------- message passing
// wave per edge; 32 lanes x 8 dims = one 256-dim row; float4 loads, f32 atomics
__global__ void scatter256_kernel(const float* __restrict__ H, const i64* __restrict__ ei,
                                  const float* __restrict__ en, float* __restrict__ OUT, long E) {
    long t = (long)blockIdx.x * blockDim.x + threadIdx.x;
    long e = t >> 5;
    int lane = threadIdx.x & 31;
    if (e >= E) return;
    i64 s = ei[e], d = ei[E + e];
    float w = en[e];
    const float4* hp = (const float4*)(H + (long)s * 256 + lane * 8);
    float4 v0 = hp[0], v1 = hp[1];
    float* op = OUT + (long)d * 256 + lane * 8;
    atomicAdd(op + 0, v0.x * w); atomicAdd(op + 1, v0.y * w);
    atomicAdd(op + 2, v0.z * w); atomicAdd(op + 3, v0.w * w);
    atomicAdd(op + 4, v1.x * w); atomicAdd(op + 5, v1.y * w);
    atomicAdd(op + 6, v1.z * w); atomicAdd(op + 7, v1.w * w);
}

// wave per edge; dims 0..39 (stride-48 padded rows, padding cols are exactly 0)
__global__ void scatter48_kernel(const float* __restrict__ H, const i64* __restrict__ ei,
                                 const float* __restrict__ en, float* __restrict__ OUT, long E) {
    long t = (long)blockIdx.x * blockDim.x + threadIdx.x;
    long e = t >> 5;
    int lane = threadIdx.x & 31;
    if (e >= E) return;
    i64 s = ei[e], d = ei[E + e];
    float w = en[e];
    float v0 = H[(long)s * 48 + lane];
    atomicAdd(OUT + (long)d * 48 + lane, v0 * w);
    if (lane < 8) {
        float v1 = H[(long)s * 48 + 32 + lane];
        atomicAdd(OUT + (long)d * 48 + 32 + lane, v1 * w);
    }
}

// AGG = relu(bn(AGG + H*self_norm + bias)) (f32, in place) and bf16 mirror for next GEMM
__global__ void finalize256_kernel(float* __restrict__ AGG, const float* __restrict__ H,
                                   const float* __restrict__ deg,
                                   const float* __restrict__ bias, const float* __restrict__ g,
                                   const float* __restrict__ beta, const float* __restrict__ m,
                                   const float* __restrict__ var,
                                   __bf16* __restrict__ outbf, long Nn) {
    long idx = (long)blockIdx.x * blockDim.x + threadIdx.x;
    if (idx >= Nn * 64) return;
    long n = idx >> 6;
    int j = (int)(idx & 63) * 4;
    float sn = 1.0f / deg[n];
    float4 agg = *(const float4*)(AGG + n * 256 + j);
    float4 h   = *(const float4*)(H   + n * 256 + j);
    float o[4] = { agg.x + h.x * sn, agg.y + h.y * sn, agg.z + h.z * sn, agg.w + h.w * sn };
    #pragma unroll
    for (int c = 0; c < 4; c++) {
        float val = o[c] + bias[j + c];
        val = (val - m[j + c]) * rsqrtf(var[j + c] + BN_EPS) * g[j + c] + beta[j + c];
        o[c] = fmaxf(val, 0.0f);
    }
    float4 r = { o[0], o[1], o[2], o[3] };
    *(float4*)(AGG + n * 256 + j) = r;
    BfPack4 p;
    p.b[0] = f2bf(o[0]); p.b[1] = f2bf(o[1]); p.b[2] = f2bf(o[2]); p.b[3] = f2bf(o[3]);
    *(uint2*)(outbf + n * 256 + j) = p.u;
}

// out[n,0:40] = log_softmax(AGG + H*self_norm + b3)
__global__ void final_kernel(const float* __restrict__ AGG, const float* __restrict__ H,
                             const float* __restrict__ deg, const float* __restrict__ b3,
                             float* __restrict__ out, long Nn) {
    long n = (long)blockIdx.x * blockDim.x + threadIdx.x;
    if (n >= Nn) return;
    float sn = 1.0f / deg[n];
    float o[40];
    float mx = -1e30f;
    #pragma unroll
    for (int j = 0; j < 40; j++) {
        o[j] = AGG[n * 48 + j] + H[n * 48 + j] * sn + b3[j];
        mx = fmaxf(mx, o[j]);
    }
    float s = 0.0f;
    #pragma unroll
    for (int j = 0; j < 40; j++) s += __expf(o[j] - mx);
    float lse = __logf(s) + mx;
    #pragma unroll
    for (int j = 0; j < 40; j++) out[n * 40 + j] = o[j] - lse;
}

// ---------------------------------------------------------------- launch
extern "C" void kernel_launch(void* const* d_in, const int* in_sizes, int n_in,
                              void* d_out, int out_size, void* d_ws, size_t ws_size,
                              hipStream_t stream) {
    (void)n_in; (void)out_size; (void)ws_size;
    const float* x     = (const float*)d_in[0];
    const i64*   ei    = (const i64*)  d_in[1];
    const float* W1    = (const float*)d_in[2];
    const float* b1    = (const float*)d_in[3];
    const float* g1    = (const float*)d_in[4];
    const float* beta1 = (const float*)d_in[5];
    const float* m1    = (const float*)d_in[6];
    const float* v1    = (const float*)d_in[7];
    const float* W2    = (const float*)d_in[8];
    const float* b2    = (const float*)d_in[9];
    const float* g2    = (const float*)d_in[10];
    const float* beta2 = (const float*)d_in[11];
    const float* m2    = (const float*)d_in[12];
    const float* v2    = (const float*)d_in[13];
    const float* W3    = (const float*)d_in[14];
    const float* b3    = (const float*)d_in[15];

    long E = (long)in_sizes[1] / 2;

    float* out = (float*)d_out;
    float* emb = out + (long)N_NODES * 40;        // node_embeddings region [N,256]

    float*  ws   = (float*)d_ws;
    float*  deg  = ws;                             // [NPAD]
    float*  en   = deg  + NPAD;                    // [E]
    float*  bufA = en   + E;                       // [NPAD*256]  pre-agg h (f32)
    float*  bufB = bufA + (long)NPAD * 256;        // [NPAD*256]  agg target / reuse as [NPAD*48]
    float*  bufC = bufB + (long)NPAD * 256;        // [NPAD*48]   layer-3 pre-agg
    __bf16* xbf  = (__bf16*)(bufC + (long)NPAD * 48);   // [NPAD*128]
    __bf16* hbf  = xbf + (long)NPAD * 128;              // [NPAD*256] bf16 activations
    __bf16* Wt   = hbf + (long)NPAD * 256;              // [256*256]  transposed weights (reused)

    const int TB = 256;
    unsigned eBlocks  = (unsigned)((E + TB - 1) / TB);
    unsigned ewBlocks = (unsigned)((E * 32 + TB - 1) / TB);
    unsigned f256Blk  = (unsigned)(((long)N_NODES * 64 + TB - 1) / TB);

    // normalization
    fill_kernel<<<2048, TB, 0, stream>>>(deg, 1.0f, (long)N_NODES);
    degree_kernel<<<eBlocks, TB, 0, stream>>>(ei, deg, E);
    edgenorm_kernel<<<eBlocks, TB, 0, stream>>>(ei, deg, en, E);

    // groups: each wave = 1 M-tile x NT N-tiles
    int g256 = (NPAD / 16) * (256 / 64);   // NT=4
    int g48  = (NPAD / 16) * (48 / 48);    // NT=3

    // layer 1: h = x @ W1 ; aggregate ; bn+relu (-> hbf)
    cvt_bf16_kernel<<<(unsigned)(((long)N_NODES * 128 / 4 + TB - 1) / TB), TB, 0, stream>>>(
        x, xbf, (long)N_NODES * 128);
    wtrans_kernel<<<(256 * 128 + TB - 1) / TB, TB, 0, stream>>>(W1, Wt, 128, 256, 256);
    gemm_wmma_kernel<128, 256, 4><<<(unsigned)((g256 + 7) / 8), TB, 0, stream>>>(
        xbf, Wt, bufA, N_NODES, g256);
    fill_kernel<<<4096, TB, 0, stream>>>(bufB, 0.0f, (long)NPAD * 256);
    scatter256_kernel<<<ewBlocks, TB, 0, stream>>>(bufA, ei, en, bufB, E);
    finalize256_kernel<<<f256Blk, TB, 0, stream>>>(bufB, bufA, deg, b1, g1, beta1, m1, v1,
                                                   hbf, (long)N_NODES);

    // layer 2: h = h1 @ W2 ; aggregate into d_out embedding region ; bn+relu (-> hbf)
    wtrans_kernel<<<(256 * 256 + TB - 1) / TB, TB, 0, stream>>>(W2, Wt, 256, 256, 256);
    gemm_wmma_kernel<256, 256, 4><<<(unsigned)((g256 + 7) / 8), TB, 0, stream>>>(
        hbf, Wt, bufA, N_NODES, g256);
    fill_kernel<<<4096, TB, 0, stream>>>(emb, 0.0f, (long)N_NODES * 256);
    scatter256_kernel<<<ewBlocks, TB, 0, stream>>>(bufA, ei, en, emb, E);
    finalize256_kernel<<<f256Blk, TB, 0, stream>>>(emb, bufA, deg, b2, g2, beta2, m2, v2,
                                                   hbf, (long)N_NODES);

    // layer 3: logits = emb @ W3 ; aggregate ; log_softmax
    wtrans_kernel<<<(48 * 256 + TB - 1) / TB, TB, 0, stream>>>(W3, Wt, 256, 40, 48);
    gemm_wmma_kernel<256, 48, 3><<<(unsigned)((g48 + 7) / 8), TB, 0, stream>>>(
        hbf, Wt, bufC, N_NODES, g48);
    fill_kernel<<<4096, TB, 0, stream>>>(bufB, 0.0f, (long)NPAD * 48);
    scatter48_kernel<<<ewBlocks, TB, 0, stream>>>(bufC, ei, en, bufB, E);
    final_kernel<<<(unsigned)((N_NODES + 127) / 128), 128, 0, stream>>>(
        bufB, bufC, deg, b3, out, (long)N_NODES);
}
